// spe_discriminator_14903536517704
// MI455X (gfx1250) — matrix operand order repaired and checked by reading, not compiled, not run on hardware
//
#include <hip/hip_runtime.h>
#include <hip/hip_bf16.h>
#include <math.h>

// ---------------------------------------------------------------------------
// Types for WMMA fragments (gfx1250, wave32)
// ---------------------------------------------------------------------------
typedef __attribute__((ext_vector_type(16))) _Float16 v16h;
typedef __attribute__((ext_vector_type(8)))  _Float16 v8h;
typedef __attribute__((ext_vector_type(8)))  float    v8f;

#define SLOPE 0.01f

// Problem constants (from reference)
#define B_      16
#define C_      150
#define HW_     1024
#define NGRID   16          // ceil(1024/64)
#define MROWS   8192        // 2 tensors * 16 * 16 * 16 rows
#define K0PAD   160         // 150 padded to multiple of 32

__device__ __forceinline__ float leaky(float x) {
    return x >= 0.0f ? x : SLOPE * x;
}

// ---------------------------------------------------------------------------
// Gather the 16x16 grid (stride 64) from both tensors, convert f32 -> f16,
// pad channels 150 -> 160 with zeros.  X layout: [MROWS, K0PAD] row-major.
// ---------------------------------------------------------------------------
__global__ void gather_convert(const float* __restrict__ real,
                               const float* __restrict__ res,
                               _Float16* __restrict__ X) {
    int idx = blockIdx.x * blockDim.x + threadIdx.x;   // kp * MROWS + m
    if (idx >= MROWS * K0PAD) return;
    int m  = idx % MROWS;
    int kp = idx / MROWS;

    _Float16 v = (_Float16)0.0f;
    if (kp < C_) {
        int t   = m >> 12;          // 0 = input_real, 1 = result
        int rem = m & 4095;
        int b   = rem >> 8;
        int i   = (rem >> 4) & 15;
        int j   = rem & 15;
        const float* src = t ? res : real;
        size_t off = ((size_t)(b * C_ + kp) << 20)      // (b,c) * 1024*1024
                   + ((size_t)(i * 64) << 10)           // h = 64*i
                   + (size_t)(j * 64);                  // w = 64*j
        v = (_Float16)src[off];
    }
    X[(size_t)m * K0PAD + kp] = v;
}

// ---------------------------------------------------------------------------
// Convert W[K, Nreal] f32 (row-major) into transposed, padded f16
// Wt[Npad][Kpad]: Wt[n*Kpad + k] = W[k*Nreal + n], zero outside K/Nreal.
// ---------------------------------------------------------------------------
__global__ void wconv(const float* __restrict__ W, _Float16* __restrict__ Wt,
                      int K, int Kpad, int Nreal, int Npad) {
    int idx = blockIdx.x * blockDim.x + threadIdx.x;
    if (idx >= Npad * Kpad) return;
    int kp = idx % Kpad;
    int n  = idx / Kpad;
    float v = 0.0f;
    if (n < Nreal && kp < K) v = W[(size_t)kp * Nreal + n];
    Wt[idx] = (_Float16)v;
}

// ---------------------------------------------------------------------------
// Fragment set for one 16x32 K-step: A chunks + NSUB B fragments.
// ---------------------------------------------------------------------------
template <int NSUB>
struct Frags {
    v8h  a0, a1;
    v16h b[NSUB];
};

template <int NSUB>
__device__ __forceinline__ void ld_frags(Frags<NSUB>& f,
                                         const _Float16* __restrict__ aPtr,
                                         const _Float16* __restrict__ bPtr,
                                         int kt, int Kpad) {
    f.a0 = *(const v8h*)(aPtr + kt);
    f.a1 = *(const v8h*)(aPtr + kt + 8);
    #pragma unroll
    for (int s = 0; s < NSUB; ++s)
        f.b[s] = *(const v16h*)(bPtr + (size_t)(s * 16) * Kpad + kt);
}

template <int NSUB>
__device__ __forceinline__ void do_wmma(v8f (&acc)[NSUB], const Frags<NSUB>& f) {
    v16h a = __builtin_shufflevector(f.a0, f.a1, 0, 1, 2, 3, 4, 5, 6, 7,
                                                 8, 9, 10, 11, 12, 13, 14, 15);
    #pragma unroll
    for (int s = 0; s < NSUB; ++s)
        acc[s] = __builtin_amdgcn_wmma_f32_16x16x32_f16(
                     false, a, false, f.b[s], (short)0, acc[s], false, false);
}

// ---------------------------------------------------------------------------
// WMMA GEMM layer: out = act(A[M,Kpad] @ Wt^T + bias)
//   A  : f16 [M, Kpad] row-major (Kpad % 32 == 0)
//   Wt : f16 [Npad, Kpad]  (transposed weights, fragment-friendly)
//   MODE 0: leaky -> f16 outH[M, Npad]
//   MODE 1: sigmoid(leaky(.)) -> f32 outF[M]  (only real column 0)
// One wave computes a 16(M) x (NSUB*16)(N) strip.  Explicit double
// buffering: loads for K-step k+1 are in flight while WMMAs consume
// K-step k, so the wait before each WMMA group is loadcnt<=6, not 0.
// All K-loop control flow is scalar (SGPR compares), so EXEC stays
// all-ones around every WMMA as the ISA requires.
//
// Fragment layouts used here (wave32; contiguous K, not the ISA's table order):
//   A 16x32 f16: lane 0-15 row M=lane, K=[0..15];
//                lane 16-31 row M=lane-16, K=[16..31]
//   B 32x16 f16: lane 0-15 col N=lane, K=[0..15];
//                lane 16-31 col N=lane-16, K=[16..31]
//   C/D 16x16 f32: VGPR r -> M = r + 8*(lane>=16), N = lane&15
// ---------------------------------------------------------------------------
template <int NSUB, int MODE>
__global__ void gemm_wmma(const _Float16* __restrict__ A,
                          const _Float16* __restrict__ Wt,
                          const float* __restrict__ bias,
                          _Float16* __restrict__ outH,
                          float* __restrict__ outF,
                          int M, int Kpad, int Npad, int nReal) {
    int lane = threadIdx.x & 31;
    // Force the wave index scalar: all per-wave control flow becomes SALU.
    int waveInBlk = __builtin_amdgcn_readfirstlane(threadIdx.x >> 5);
    int wave = blockIdx.x * (blockDim.x >> 5) + waveInBlk;

    int nGroups = Npad / (NSUB * 16);
    int mTile   = wave / nGroups;
    if (mTile >= (M >> 4)) return;          // uniform scalar branch
    int nBase = (wave % nGroups) * (NSUB * 16);

    int half = lane >> 4;      // 0: lanes 0-15, 1: lanes 16-31
    int l15  = lane & 15;

    // Per-lane base pointers (lane-half K-chunk selection folded in).
    const _Float16* aPtr = A + (size_t)(mTile * 16 + l15) * Kpad + 16 * half;
    const _Float16* bPtr = Wt + (size_t)(nBase + l15) * Kpad + 16 * half;

    v8f acc[NSUB];
    #pragma unroll
    for (int s = 0; s < NSUB; ++s) acc[s] = v8f{};

    // Software-pipelined K loop with two rotating fragment sets.
    Frags<NSUB> f0, f1;
    ld_frags<NSUB>(f0, aPtr, bPtr, 0, Kpad);
    int kt = 0;
    for (;;) {
        int kn = kt + 32;
        if (kn < Kpad) ld_frags<NSUB>(f1, aPtr, bPtr, kn, Kpad);
        do_wmma<NSUB>(acc, f0);
        kt = kn;
        if (kt >= Kpad) break;

        kn = kt + 32;
        if (kn < Kpad) ld_frags<NSUB>(f0, aPtr, bPtr, kn, Kpad);
        do_wmma<NSUB>(acc, f1);
        kt = kn;
        if (kt >= Kpad) break;
    }

    // Epilogue: bias + activation + store per C/D layout
    #pragma unroll
    for (int s = 0; s < NSUB; ++s) {
        int col = nBase + s * 16 + l15;
        float bv = (col < nReal) ? bias[col] : 0.0f;
        #pragma unroll
        for (int r = 0; r < 8; ++r) {
            int row = mTile * 16 + half * 8 + r;
            float v = leaky(acc[s][r] + bv);
            if (MODE == 0) {
                outH[(size_t)row * Npad + col] = (_Float16)v;
            } else if (col == 0) {
                outF[row] = 1.0f / (1.0f + expf(-v));
            }
        }
    }
}

// ---------------------------------------------------------------------------
// Host-side orchestration
// ---------------------------------------------------------------------------
extern "C" void kernel_launch(void* const* d_in, const int* in_sizes, int n_in,
                              void* d_out, int out_size, void* d_ws, size_t ws_size,
                              hipStream_t stream) {
    (void)in_sizes; (void)n_in; (void)out_size; (void)ws_size;

    const float* inp_real = (const float*)d_in[0];
    const float* inp_res  = (const float*)d_in[1];
    const float* W[6]; const float* bia[6];
    for (int i = 0; i < 6; ++i) {
        W[i]   = (const float*)d_in[2 + 2 * i];
        bia[i] = (const float*)d_in[3 + 2 * i];
    }
    float* out = (float*)d_out;

    // Layer geometry: K, Kpad, Nreal, Npad
    const int  Ks[6]    = {150, 128, 256, 512, 1024, 512};
    const int  Kpads[6] = {160, 128, 256, 512, 1024, 512};
    const int  Nrs[6]   = {128, 256, 512, 1024, 512, 1};
    const int  Npads[6] = {128, 256, 512, 1024, 512, 16};

    // Workspace layout
    char* ws = (char*)d_ws;
    _Float16* act0 = (_Float16*)ws;                         // 16 MB
    _Float16* act1 = (_Float16*)(ws + ((size_t)16 << 20));  // 16 MB
    size_t off = (size_t)32 << 20;
    _Float16* wt[6];
    for (int i = 0; i < 6; ++i) {
        wt[i] = (_Float16*)(ws + off);
        off += (size_t)Npads[i] * Kpads[i] * sizeof(_Float16);
    }

    // 1) Gather + f32->f16 convert into act0 [MROWS, 160]
    {
        int total  = MROWS * K0PAD;
        int blocks = (total + 255) / 256;
        gather_convert<<<blocks, 256, 0, stream>>>(inp_real, inp_res, act0);
    }

    // 2) Convert & transpose weights into fragment-friendly f16
    for (int i = 0; i < 6; ++i) {
        int total  = Npads[i] * Kpads[i];
        int blocks = (total + 255) / 256;
        wconv<<<blocks, 256, 0, stream>>>(W[i], wt[i], Ks[i], Kpads[i],
                                          Nrs[i], Npads[i]);
    }

    // 3) Six WMMA GEMM layers (ping-pong act0/act1, last layer -> d_out)
    _Float16* bufs[2] = {act0, act1};
    int cur = 0;
    for (int i = 0; i < 6; ++i) {
        int Kpad = Kpads[i];
        int Npad = Npads[i];
        _Float16* dstH = bufs[cur ^ 1];

        if (i < 5) {
            // Npad is a multiple of 64: 16x64 strip per wave, NSUB=4
            int nGroups = Npad / 64;
            int waves   = (MROWS >> 4) * nGroups;
            int blocks  = (waves * 32 + 255) / 256;
            gemm_wmma<4, 0><<<blocks, 256, 0, stream>>>(
                bufs[cur], wt[i], bia[i], dstH, out,
                MROWS, Kpad, Npad, Nrs[i]);
        } else {
            // Final layer: Npad=16, NSUB=1, sigmoid -> d_out (f32)
            int waves  = (MROWS >> 4);
            int blocks = (waves * 32 + 255) / 256;
            gemm_wmma<1, 1><<<blocks, 256, 0, stream>>>(
                bufs[cur], wt[i], bia[i], dstH, out,
                MROWS, Kpad, Npad, Nrs[i]);
        }
        cur ^= 1;
    }
}
